// SelfAttention_8452495638621
// MI455X (gfx1250) — matrix operand order repaired and checked by reading, not compiled, not run on hardware
//
#include <hip/hip_runtime.h>

typedef __attribute__((ext_vector_type(16))) _Float16 v16h;
typedef __attribute__((ext_vector_type(8)))  float    v8f;

namespace {

constexpr int BB = 4;
constexpr int NN = 2048;
constexpr int CC = 768;   // hidden
constexpr int HH = 12;
constexpr int DD = 64;    // d_k == d_v

union AF { v16h v; _Float16 h[16]; };
union CF { v8f  v; float    f[8];  };

__device__ inline v8f wmma16(v16h a, v16h b, v8f c) {
  // D = A(16x32,f16) * B(32x16,f16) + C(16x16,f32)
  return __builtin_amdgcn_wmma_f32_16x16x32_f16(
      /*neg_a=*/false, a, /*neg_b=*/false, b,
      /*c_mod=*/(short)0, c, /*reuse_a=*/false, /*reuse_b=*/false);
}

// A-fragment from row-major matrix: element [m][k], rows contiguous (ld elems).
// lane L: row = m0 + (L&15); halves 0..7 -> k0+kb+e, halves 8..15 -> k0+kb+16+e,
// kb = (L<16)?0:8  (ISA 16-bit A 16x32 layout).
__device__ inline v16h load_a_rm(const _Float16* p, int ld, int m0, int k0) {
  const int lane = threadIdx.x & 31;
  AF f;
  const int row = m0 + (lane & 15);
  const int kb  = (lane & 16) ? 8 : 0;
  const _Float16* r = p + (size_t)row * ld + k0 + kb;
#pragma unroll
  for (int e = 0; e < 8; ++e) f.h[e] = r[e];
#pragma unroll
  for (int e = 0; e < 8; ++e) f.h[8 + e] = r[16 + e];
  return f.v;
}

// B-fragment of W^T where W is row-major (ld elems): B[k][n] = W[n][k].
// lane L: col = n0 + (L&15); halves 0..7 -> k0+kb+e, halves 8..15 -> k0+kb+16+e, kb=(L<16)?0:8.
__device__ inline v16h load_bT(const _Float16* p, int ld, int n0, int k0) {
  const int lane = threadIdx.x & 31;
  AF f;
  const int col = n0 + (lane & 15);
  const int kb  = (lane & 16) ? 8 : 0;
  const _Float16* r = p + (size_t)col * ld + k0 + kb;
#pragma unroll
  for (int e = 0; e < 8; ++e) f.h[e] = r[e];
#pragma unroll
  for (int e = 0; e < 8; ++e) f.h[8 + e] = r[16 + e];
  return f.v;
}

__device__ inline float rowmax16(float x) {
#pragma unroll
  for (int m = 1; m < 16; m <<= 1) x = fmaxf(x, __shfl_xor(x, m, 32));
  return x;
}
__device__ inline float rowsum16(float x) {
#pragma unroll
  for (int m = 1; m < 16; m <<= 1) x += __shfl_xor(x, m, 32);
  return x;
}

// ---------------- kernel 0: transpose x -> (B,N,C) f16; convert weights ------
__global__ __launch_bounds__(256)
void convert_kernel(const float* __restrict__ x,  const float* __restrict__ Wq,
                    const float* __restrict__ Wk, const float* __restrict__ Wv,
                    const float* __restrict__ Wo,
                    _Float16* __restrict__ xt,  _Float16* __restrict__ wq,
                    _Float16* __restrict__ wk,  _Float16* __restrict__ wv,
                    _Float16* __restrict__ wo) {
  const long XT = (long)BB * NN * CC;
  const long W1 = (long)HH * DD * CC;
  const long WO = (long)CC * CC;
  const long total = XT + 3 * W1 + WO;
  for (long i = (long)blockIdx.x * blockDim.x + threadIdx.x; i < total;
       i += (long)gridDim.x * blockDim.x) {
    if (i < XT) {
      long c = i % CC; long t = i / CC;
      long n = t % NN; long b = t / NN;
      xt[i] = (_Float16)x[(b * CC + c) * NN + n];
    } else if (i < XT + W1) {
      long j = i - XT;          wq[j] = (_Float16)Wq[j];
    } else if (i < XT + 2 * W1) {
      long j = i - XT - W1;     wk[j] = (_Float16)Wk[j];
    } else if (i < XT + 3 * W1) {
      long j = i - XT - 2 * W1; wv[j] = (_Float16)Wv[j];
    } else {
      long j = i - XT - 3 * W1; wo[j] = (_Float16)Wo[j];
    }
  }
}

// ---------------- kernel 1: Q/K/V projections ---------------------------------
// One wave computes a 16(n) x 64(d) slab: A-fragment reused across 4 B-tiles.
// Q,K stored (B,H,N,64) row-major; V stored TRANSPOSED (B,H,64,N) so the
// attention P@V B-fragment reads contiguous 8-half runs.
__global__ __launch_bounds__(256)
void qkv_kernel(const _Float16* __restrict__ xt,
                const _Float16* __restrict__ wq, const _Float16* __restrict__ wk,
                const _Float16* __restrict__ wv,
                _Float16* __restrict__ q, _Float16* __restrict__ k,
                _Float16* __restrict__ v) {
  const int lane = threadIdx.x & 31;
  int wave = (blockIdx.x * blockDim.x + threadIdx.x) >> 5;
  const int NT = NN / 16;                               // 128
  int t  = wave;
  int nt = t % NT; t /= NT;
  int h  = t % HH; t /= HH;
  int b  = t % BB; t /= BB;
  int which = t;                                        // 0=q 1=k 2=v
  const _Float16* wsel = (which == 0) ? wq : (which == 1) ? wk : wv;
  _Float16*       osel = (which == 0) ? q  : (which == 1) ? k  : v;

  const _Float16* xrow = xt + (size_t)b * NN * CC;      // (N,C) row-major
  const _Float16* wmat = wsel + (size_t)h * DD * CC;    // (D,C) row-major
  const int n0 = nt * 16;

  CF a0, a1, a2, a3;
  a0.v = v8f{}; a1.v = v8f{}; a2.v = v8f{}; a3.v = v8f{};
#pragma unroll 2
  for (int k0 = 0; k0 < CC; k0 += 32) {
    v16h a = load_a_rm(xrow, CC, n0, k0);               // reused 4x
    a0.v = wmma16(a, load_bT(wmat, CC, 0,  k0), a0.v);
    a1.v = wmma16(a, load_bT(wmat, CC, 16, k0), a1.v);
    a2.v = wmma16(a, load_bT(wmat, CC, 32, k0), a2.v);
    a3.v = wmma16(a, load_bT(wmat, CC, 48, k0), a3.v);
  }

  const int cl = lane & 15;
  const int rb = (lane >> 4) * 8;
  _Float16* out = osel + (size_t)(b * HH + h) * NN * DD;
  if (which == 2) {
    // transposed store: Vt[d, n]
#pragma unroll
    for (int vv = 0; vv < 8; ++vv) {
      int n = n0 + rb + vv;
      out[(size_t)(0  + cl) * NN + n] = (_Float16)a0.f[vv];
      out[(size_t)(16 + cl) * NN + n] = (_Float16)a1.f[vv];
      out[(size_t)(32 + cl) * NN + n] = (_Float16)a2.f[vv];
      out[(size_t)(48 + cl) * NN + n] = (_Float16)a3.f[vv];
    }
  } else {
#pragma unroll
    for (int vv = 0; vv < 8; ++vv) {
      size_t rbase = (size_t)(n0 + rb + vv) * DD;
      out[rbase + 0  + cl] = (_Float16)a0.f[vv];
      out[rbase + 16 + cl] = (_Float16)a1.f[vv];
      out[rbase + 32 + cl] = (_Float16)a2.f[vv];
      out[rbase + 48 + cl] = (_Float16)a3.f[vv];
    }
  }
}

// ---------------- kernel 2: streaming-softmax attention ----------------------
// One wave per (b, h, 16-query tile). Keys streamed in chunks of 32.
// Q,K are (N,64) row-major; V is (64,N) (transposed).
__global__ __launch_bounds__(128)
void attn_kernel(const _Float16* __restrict__ q, const _Float16* __restrict__ k,
                 const _Float16* __restrict__ v, const int* __restrict__ mask,
                 _Float16* __restrict__ comb) {
  constexpr int PSTR = 40;                 // 80B row stride: 16B-aligned runs
  __shared__ _Float16 plds[4][16 * PSTR];
  const int lane   = threadIdx.x & 31;
  const int wavein = threadIdx.x >> 5;
  int wave = blockIdx.x * (blockDim.x >> 5) + wavein;
  const int NT = NN / 16;
  int t  = wave;
  int it = t % NT; t /= NT;
  int h  = t % HH; t /= HH;
  int b  = t;

  const size_t hoff = (size_t)(b * HH + h) * NN * DD;
  const _Float16* Q  = q + hoff;
  const _Float16* K  = k + hoff;
  const _Float16* Vt = v + hoff;           // (64, N)
  const int i0   = it * 16;
  const int colj = lane & 15;
  const int rb   = (lane >> 4) * 8;
  const float scale = 0.125f;              // 1/sqrt(64)

  // Q tile (16 x 64) as two A-fragments, kept in VGPRs for the whole loop
  v16h aq0 = load_a_rm(Q, DD, i0, 0);
  v16h aq1 = load_a_rm(Q, DD, i0, 32);

  CF acc0, acc1, acc2, acc3;
  acc0.v = v8f{}; acc1.v = v8f{}; acc2.v = v8f{}; acc3.v = v8f{};
  float mrow[8], lrow[8];
#pragma unroll
  for (int vv = 0; vv < 8; ++vv) { mrow[vv] = -__builtin_inff(); lrow[vv] = 0.f; }

  _Float16* pl = plds[wavein];
  const int* mrow_ptr = mask + (size_t)b * NN;

  for (int j0 = 0; j0 < NN; j0 += 32) {
    // scores S0 (cols j0..j0+15), S1 (cols j0+16..j0+31)
    CF s0, s1; s0.v = v8f{}; s1.v = v8f{};
    s0.v = wmma16(aq0, load_bT(K, DD, j0,      0),  s0.v);
    s0.v = wmma16(aq1, load_bT(K, DD, j0,      32), s0.v);
    s1.v = wmma16(aq0, load_bT(K, DD, j0 + 16, 0),  s1.v);
    s1.v = wmma16(aq1, load_bT(K, DD, j0 + 16, 32), s1.v);

    const int mk0 = mrow_ptr[j0 + colj];
    const int mk1 = mrow_ptr[j0 + 16 + colj];

    float p0[8], p1[8];
#pragma unroll
    for (int vv = 0; vv < 8; ++vv) {
      float x0 = mk0 ? s0.f[vv] * scale : -__builtin_inff();
      float x1 = mk1 ? s1.f[vv] * scale : -__builtin_inff();
      p0[vv] = x0; p1[vv] = x1;
      float tmax = rowmax16(fmaxf(x0, x1));
      float mn   = fmaxf(mrow[vv], tmax);
      float alpha = (mrow[vv] == -__builtin_inff()) ? 0.f : __expf(mrow[vv] - mn);
      float e0 = (p0[vv] == -__builtin_inff()) ? 0.f : __expf(p0[vv] - mn);
      float e1 = (p1[vv] == -__builtin_inff()) ? 0.f : __expf(p1[vv] - mn);
      p0[vv] = e0; p1[vv] = e1;
      lrow[vv] = lrow[vv] * alpha + rowsum16(e0 + e1);
      mrow[vv] = mn;
      acc0.f[vv] *= alpha; acc1.f[vv] *= alpha;
      acc2.f[vv] *= alpha; acc3.f[vv] *= alpha;
    }

    // stage P (16 x 32, f16) through LDS to re-shape C-layout -> A-layout
#pragma unroll
    for (int vv = 0; vv < 8; ++vv) {
      pl[(rb + vv) * PSTR + colj]      = (_Float16)p0[vv];
      pl[(rb + vv) * PSTR + 16 + colj] = (_Float16)p1[vv];
    }
    __syncthreads();
    v16h ap = load_a_rm(pl, PSTR, 0, 0);
    __syncthreads();

    // ctx += P @ V : B-fragment [j][d] = Vt[d][j], contiguous over j
    acc0.v = wmma16(ap, load_bT(Vt, NN, 0,  j0), acc0.v);
    acc1.v = wmma16(ap, load_bT(Vt, NN, 16, j0), acc1.v);
    acc2.v = wmma16(ap, load_bT(Vt, NN, 32, j0), acc2.v);
    acc3.v = wmma16(ap, load_bT(Vt, NN, 48, j0), acc3.v);
  }

  // normalize + write combined[b, n, h*64 + d] (f16, GEMM-ready)
#pragma unroll
  for (int vv = 0; vv < 8; ++vv) {
    int row = i0 + rb + vv;
    float inv = (lrow[vv] > 0.f) ? 1.f / lrow[vv] : 0.f;
    size_t base = ((size_t)b * NN + row) * (HH * DD) + (size_t)h * DD;
    comb[base + 0  + colj] = (_Float16)(acc0.f[vv] * inv);
    comb[base + 16 + colj] = (_Float16)(acc1.f[vv] * inv);
    comb[base + 32 + colj] = (_Float16)(acc2.f[vv] * inv);
    comb[base + 48 + colj] = (_Float16)(acc3.f[vv] * inv);
  }
}

// ---------------- kernel 3: out[b,c,n] = combined @ Wo^T ---------------------
// One wave computes 16(n) x 64(cout): A-fragment reused across 4 B-tiles.
__global__ __launch_bounds__(256)
void proj_kernel(const _Float16* __restrict__ comb,
                 const _Float16* __restrict__ wo, float* __restrict__ out) {
  const int lane = threadIdx.x & 31;
  int wave = (blockIdx.x * blockDim.x + threadIdx.x) >> 5;
  const int CT4 = CC / 64, NT = NN / 16;                // 12, 128
  int t  = wave;
  int ct = t % CT4; t /= CT4;
  int nt = t % NT;  t /= NT;
  int b  = t;

  const _Float16* A = comb + (size_t)b * NN * CC;       // (N, 768) row-major
  const int n0 = nt * 16, c0 = ct * 64;
  CF a0, a1, a2, a3;
  a0.v = v8f{}; a1.v = v8f{}; a2.v = v8f{}; a3.v = v8f{};
#pragma unroll 2
  for (int k0 = 0; k0 < CC; k0 += 32) {
    v16h a = load_a_rm(A, CC, n0, k0);                  // reused 4x
    a0.v = wmma16(a, load_bT(wo, CC, c0,      k0), a0.v);
    a1.v = wmma16(a, load_bT(wo, CC, c0 + 16, k0), a1.v);
    a2.v = wmma16(a, load_bT(wo, CC, c0 + 32, k0), a2.v);
    a3.v = wmma16(a, load_bT(wo, CC, c0 + 48, k0), a3.v);
  }
  const int cl = lane & 15;
  const int rb = (lane >> 4) * 8;
  float* O = out + (size_t)b * CC * NN;
#pragma unroll
  for (int vv = 0; vv < 8; ++vv) {
    int n = n0 + rb + vv;
    O[(size_t)(c0 + 0  + cl) * NN + n] = a0.f[vv];
    O[(size_t)(c0 + 16 + cl) * NN + n] = a1.f[vv];
    O[(size_t)(c0 + 32 + cl) * NN + n] = a2.f[vv];
    O[(size_t)(c0 + 48 + cl) * NN + n] = a3.f[vv];
  }
}

} // anonymous namespace

extern "C" void kernel_launch(void* const* d_in, const int* in_sizes, int n_in,
                              void* d_out, int out_size, void* d_ws, size_t ws_size,
                              hipStream_t stream) {
  const float* x  = (const float*)d_in[0];
  const float* Wq = (const float*)d_in[1];
  const float* Wk = (const float*)d_in[2];
  const float* Wv = (const float*)d_in[3];
  const float* Wo = (const float*)d_in[4];
  const int*   mk = (const int*)  d_in[5];
  float* out = (float*)d_out;

  const size_t XT = (size_t)BB * NN * CC;      // 6,291,456
  const size_t W1 = (size_t)HH * DD * CC;      //   589,824
  const size_t WO = (size_t)CC * CC;           //   589,824
  const size_t QK = (size_t)BB * HH * NN * DD; // 6,291,456

  _Float16* ws = (_Float16*)d_ws;
  size_t off = 0;
  _Float16* xt16 = ws + off; off += XT;
  _Float16* wq16 = ws + off; off += W1;
  _Float16* wk16 = ws + off; off += W1;
  _Float16* wv16 = ws + off; off += W1;
  _Float16* wo16 = ws + off; off += WO;
  _Float16* q16  = ws + off; off += QK;
  _Float16* k16  = ws + off; off += QK;
  _Float16* v16  = ws + off; off += QK;        // stored (B,H,64,N)
  _Float16* cmb  = ws + off; off += XT;        // total ~67.6 MB of f16

  // 0) transpose x to (B,N,C) f16 + convert all weights to f16
  convert_kernel<<<4096, 256, 0, stream>>>(x, Wq, Wk, Wv, Wo,
                                           xt16, wq16, wk16, wv16, wo16);

  // 1) QKV projections: 3*B*H*(N/16) = 18432 waves, 16x64 slab each
  qkv_kernel<<<18432 * 32 / 256, 256, 0, stream>>>(xt16, wq16, wk16, wv16,
                                                   q16, k16, v16);

  // 2) streaming-softmax attention: B*H*(N/16) = 6144 waves, 4 waves/block
  attn_kernel<<<6144 * 32 / 128, 128, 0, stream>>>(q16, k16, v16, mk, cmb);

  // 3) output projection: B*(N/16)*(C/64) = 6144 waves, 16x64 slab each
  proj_kernel<<<6144 * 32 / 256, 256, 0, stream>>>(cmb, wo16, out);
}
